// OpsCell_79852031967861
// MI455X (gfx1250) — matrix operand order repaired and checked by reading, not compiled, not run on hardware
//
#include <hip/hip_runtime.h>

#define OBS 128
#define ACT 32
#define HID 64
#define TT  128
#define BB  4096
#define TB  (TT*BB)

typedef __attribute__((ext_vector_type(16))) __bf16 v16bf;
typedef __attribute__((ext_vector_type(8)))  __bf16 v8bf;
typedef __attribute__((ext_vector_type(4)))  __bf16 v4bf;
typedef __attribute__((ext_vector_type(8)))  float  v8f;
typedef __attribute__((ext_vector_type(4)))  float  v4f;

__device__ __forceinline__ v8f wmma_bf16(v16bf a, v16bf b, v8f c) {
  // D = A(16x32 bf16) * B(32x16 bf16) + C(16x16 f32)
  return __builtin_amdgcn_wmma_f32_16x16x32_bf16(false, a, false, b, (short)0, c,
                                                 false, false);
}

// Pack 16 fp32 values (4x v4f) into a 16x32 bf16 A-fragment register.
__device__ __forceinline__ v16bf pack_afrag(v4f a0, v4f a1, v4f a2, v4f a3) {
  v16bf r;
#pragma unroll
  for (int i = 0; i < 4; ++i) {
    r[i]      = (__bf16)a0[i];
    r[4 + i]  = (__bf16)a1[i];
    r[8 + i]  = (__bf16)a2[i];
    r[12 + i] = (__bf16)a3[i];
  }
  return r;
}

// Non-temporal A-fragment load from fp32 global memory.
// p -> 8 contiguous floats for halves 0..7, p+16 -> 8 floats for halves 8..15.
__device__ __forceinline__ v16bf load_afrag_nt(const float* __restrict__ p) {
  v4f a0 = __builtin_nontemporal_load((const v4f*)p);
  v4f a1 = __builtin_nontemporal_load((const v4f*)(p + 4));
  v4f a2 = __builtin_nontemporal_load((const v4f*)(p + 16));
  v4f a3 = __builtin_nontemporal_load((const v4f*)(p + 20));
  return pack_afrag(a0, a1, a2, a3);
}

// Read a 16x32 A-fragment (K block kb) from a row-major [16][64] bf16 LDS tile.
__device__ __forceinline__ v16bf lds_afrag(const __bf16* tile, int lane, int kb) {
  const int hk = (lane & 16) ? 8 : 0;
  const __bf16* tp = tile + (lane & 15) * 64 + kb * 32 + hk;
  v8bf lo = *(const v8bf*)tp;         // ds_read_b128
  v8bf hi = *(const v8bf*)(tp + 16);  // ds_read_b128
  return __builtin_shufflevector(lo, hi, 0, 1, 2, 3, 4, 5, 6, 7,
                                 8, 9, 10, 11, 12, 13, 14, 15);
}

__device__ __forceinline__ v4bf cvt4_bf16(v4f v) {
  v4bf r;
#pragma unroll
  for (int i = 0; i < 4; ++i) r[i] = (__bf16)v[i];
  return r;
}

// ---------------------------------------------------------------------------
// Kernel 1: capture path  z1 = tanh(tanh(x Wc1 + bc1) Wc2 + bc2), all T*B rows.
// Persistent grid; weights pre-swizzled into WMMA B-fragment layout in LDS.
// Result staged in LDS so capt stores are fully coalesced global_store_b128.
// ---------------------------------------------------------------------------
__global__ void __launch_bounds__(256) capture_kernel(
    const float* __restrict__ x,
    const float* __restrict__ Wc1, const float* __restrict__ bc1,
    const float* __restrict__ Wc2, const float* __restrict__ bc2,
    float* __restrict__ capt)
{
  __shared__ __align__(16) __bf16 sW1[16 * 512];   // 4 kb x 4 nt B-fragments
  __shared__ __align__(16) __bf16 sW2[8 * 512];    // 2 kb x 4 nt B-fragments
  __shared__ __align__(16) float  sScr[8][16 * 64]; // per-wave scratch (bf16/f32 union)

  const int tid = threadIdx.x;

  // Stage weights as bf16 B-fragments: addr = frag*512 + lane*16 + j
  // lane<16 : col = nt*16+lane,    K = kb*32 + (j<8 ? j : 8+j)
  // lane>=16: col = nt*16+lane-16, K = kb*32 + 8 + (j<8 ? j : 8+j)
  for (int i = tid; i < 16 * 512; i += 256) {
    int f = i >> 9, l = (i >> 4) & 31, j = i & 15;
    int kb = f >> 2, nt = f & 3;
    int k = kb * 32 + ((l & 16) ? 8 : 0) + (j < 8 ? j : j + 8);
    sW1[i] = (__bf16)Wc1[k * HID + nt * 16 + (l & 15)];
  }
  for (int i = tid; i < 8 * 512; i += 256) {
    int f = i >> 9, l = (i >> 4) & 31, j = i & 15;
    int kb = f >> 2, nt = f & 3;
    int k = kb * 32 + ((l & 16) ? 8 : 0) + (j < 8 ? j : j + 8);
    sW2[i] = (__bf16)Wc2[k * HID + nt * 16 + (l & 15)];
  }
  __syncthreads();

  const int lane = tid & 31;
  const int wave = tid >> 5;
  const int cl = lane & 15;
  const int hk = (lane & 16) ? 8 : 0;   // C-row offset == A-load K sub-offset
  float*  zT = &sScr[wave][0];          // fp32 [16][64] staging view
  __bf16* tT = (__bf16*)zT;             // bf16 [16][64] bounce view

  const int tilesTotal = TB / 16;       // 32768 M-tiles of 16 rows
  const int stride = gridDim.x * 8;     // waves in grid

  for (int mt = blockIdx.x * 8 + wave; mt < tilesTotal; mt += stride) {
    const long rowbase = (long)mt * 16;

    // GEMM1: [16,128] x [128,64], K in 4 blocks of 32 (x streamed non-temporal)
    v8f acc[4];
#pragma unroll
    for (int nt = 0; nt < 4; ++nt) acc[nt] = (v8f){};
#pragma unroll
    for (int kb = 0; kb < 4; ++kb) {
      v16bf af = load_afrag_nt(x + (rowbase + cl) * OBS + kb * 32 + hk);
#pragma unroll
      for (int nt = 0; nt < 4; ++nt) {
        v16bf bf = *(const v16bf*)(&sW1[(kb * 4 + nt) * 512 + lane * 16]);
        acc[nt] = wmma_bf16(af, bf, acc[nt]);
      }
    }

    // bias + tanh, C-layout -> row-major bf16 bounce tile
#pragma unroll
    for (int nt = 0; nt < 4; ++nt) {
      const int col = nt * 16 + cl;
      const float bias = bc1[col];
#pragma unroll
      for (int r = 0; r < 8; ++r)
        tT[(hk + r) * 64 + col] = (__bf16)tanhf(acc[nt][r] + bias);
    }

    // GEMM2: [16,64] x [64,64]
    v8f acc2[4];
#pragma unroll
    for (int nt = 0; nt < 4; ++nt) acc2[nt] = (v8f){};
#pragma unroll
    for (int kb = 0; kb < 2; ++kb) {
      v16bf af = lds_afrag(tT, lane, kb);
#pragma unroll
      for (int nt = 0; nt < 4; ++nt) {
        v16bf bf = *(const v16bf*)(&sW2[(kb * 4 + nt) * 512 + lane * 16]);
        acc2[nt] = wmma_bf16(af, bf, acc2[nt]);
      }
    }

    // bias + tanh -> fp32 staging tile (WAR on tT protected by data dep)
#pragma unroll
    for (int nt = 0; nt < 4; ++nt) {
      const int col = nt * 16 + cl;
      const float bias = bc2[col];
#pragma unroll
      for (int r = 0; r < 8; ++r)
        zT[(hk + r) * 64 + col] = tanhf(acc2[nt][r] + bias);
    }

    // Coalesced emit: lane <-> 16B chunk, 512B contiguous per instruction.
    float* cp = capt + rowbase * 64;
#pragma unroll
    for (int j = 0; j < 8; ++j) {
      const int fi = (j * 32 + lane) * 4;
      v4f v = *(const v4f*)(zT + fi);   // ds_read_b128
      *(v4f*)(cp + fi) = v;             // global_store_b128, fully coalesced
    }
    asm volatile("" ::: "memory");      // keep LDS reuse ordered across iterations
  }
}

// ---------------------------------------------------------------------------
// Kernel 2: sequential scan over T. One 16-row batch tile per wave; Wp1/Wp2
// B-fragments live in registers (reused 128x), h lives in a wave-private
// LDS tile. z1/g/a loads hoisted + next-step prefetch to shorten the serial
// critical path; all global traffic is coalesced b128.
// ---------------------------------------------------------------------------
__global__ void __launch_bounds__(256) scan_kernel(
    const float* __restrict__ h0, const float* __restrict__ g,
    const float* __restrict__ a,
    const float* __restrict__ Wp1, const float* __restrict__ bp1,
    const float* __restrict__ Wp2, const float* __restrict__ bp2,
    const float* __restrict__ capt,
    float* __restrict__ outs, float* __restrict__ hlast)
{
  __shared__ __align__(16) __bf16 sH[8][16 * 64];    // per-wave h tile (bf16)
  __shared__ __align__(16) float  sScr[8][16 * 64];  // per-wave scratch (bf16/f32 union)

  const int tid = threadIdx.x;
  const int lane = tid & 31;
  const int wave = tid >> 5;
  const int cl = lane & 15;
  const int hk = (lane & 16) ? 8 : 0;
  const long rowb = (long)(blockIdx.x * 8 + wave) * 16;  // batch row base

  // Weight B-fragments straight into registers (coalesced over the 16 cols).
  v16bf wB1[12], wB2[8];
#pragma unroll
  for (int f = 0; f < 12; ++f) {                 // Wp1: 3 K-blocks x 4 N-tiles
    const int kb = f >> 2, nt = f & 3;
    const int col = nt * 16 + cl;
    const int kbase = kb * 32 + ((lane & 16) ? 8 : 0);
    v16bf w;
#pragma unroll
    for (int j = 0; j < 16; ++j) w[j] = (__bf16)Wp1[(kbase + (j < 8 ? j : j + 8)) * HID + col];
    wB1[f] = w;
  }
#pragma unroll
  for (int f = 0; f < 8; ++f) {                  // Wp2: 2 K-blocks x 4 N-tiles
    const int kb = f >> 2, nt = f & 3;
    const int col = nt * 16 + cl;
    const int kbase = kb * 32 + ((lane & 16) ? 8 : 0);
    v16bf w;
#pragma unroll
    for (int j = 0; j < 16; ++j) w[j] = (__bf16)Wp2[(kbase + (j < 8 ? j : j + 8)) * HID + col];
    wB2[f] = w;
  }

  float b1[4], b2[4];
#pragma unroll
  for (int nt = 0; nt < 4; ++nt) {
    b1[nt] = bp1[nt * 16 + cl];
    b2[nt] = bp2[nt * 16 + cl];
  }

  float*  zT = &sScr[wave][0];
  __bf16* tT = (__bf16*)zT;
  __bf16* hT = &sH[wave][0];

  // h tile init (row-major [16][64] == 16 contiguous h0 rows)
  for (int i = lane; i < 16 * 64; i += 32)
    hT[i] = (__bf16)h0[rowb * 64 + i];

  for (int t = 0; t < TT; ++t) {
    const long tb = (long)t * BB;
    const float* z1p = capt + (tb + rowb) * 64;  // contiguous 4KB tile
    float*       op  = outs + (tb + rowb) * 64;

    // --- hoisted h-independent loads (hide latency behind the GEMM chain) ---
    v4f z1r[8];
    float gv[8];
#pragma unroll
    for (int j = 0; j < 8; ++j) {
      z1r[j] = *(const v4f*)(z1p + (j * 32 + lane) * 4);  // L2-resident (RT)
      gv[j]  = __builtin_nontemporal_load(g + tb + rowb + 2 * j + (lane >> 4));
    }
    v16bf afA = load_afrag_nt(a + (tb + rowb + cl) * ACT + hk);

    // prefetch next step's capt / a tiles while this step computes
    if (t + 1 < TT) {
      const long tbn = tb + BB;
      __builtin_prefetch(capt + (tbn + rowb) * 64 + lane * 32, 0, 3);
      __builtin_prefetch(a + (tbn + rowb) * ACT + lane * 16, 0, 0);
    }

    // --- GEMM1: [h | a] (16x96) x Wp1 (96x64) ---
    v8f acc[4];
#pragma unroll
    for (int nt = 0; nt < 4; ++nt) acc[nt] = (v8f){};
#pragma unroll
    for (int kb = 0; kb < 2; ++kb) {             // K 0..63 from h tile
      v16bf af = lds_afrag(hT, lane, kb);
#pragma unroll
      for (int nt = 0; nt < 4; ++nt) acc[nt] = wmma_bf16(af, wB1[kb * 4 + nt], acc[nt]);
    }
#pragma unroll
    for (int nt = 0; nt < 4; ++nt) acc[nt] = wmma_bf16(afA, wB1[8 + nt], acc[nt]);

    // bias + tanh -> bounce tile
#pragma unroll
    for (int nt = 0; nt < 4; ++nt) {
      const int col = nt * 16 + cl;
#pragma unroll
      for (int r = 0; r < 8; ++r)
        tT[(hk + r) * 64 + col] = (__bf16)tanhf(acc[nt][r] + b1[nt]);
    }

    // --- GEMM2: (16x64) x Wp2 (64x64) ---
    v8f acc2[4];
#pragma unroll
    for (int nt = 0; nt < 4; ++nt) acc2[nt] = (v8f){};
#pragma unroll
    for (int kb = 0; kb < 2; ++kb) {
      v16bf af = lds_afrag(tT, lane, kb);
#pragma unroll
      for (int nt = 0; nt < 4; ++nt) acc2[nt] = wmma_bf16(af, wB2[kb * 4 + nt], acc2[nt]);
    }

    // bias + tanh -> fp32 staging tile (WAR on tT protected by data dep)
#pragma unroll
    for (int nt = 0; nt < 4; ++nt) {
      const int col = nt * 16 + cl;
#pragma unroll
      for (int r = 0; r < 8; ++r)
        zT[(hk + r) * 64 + col] = tanhf(acc2[nt][r] + b2[nt]);
    }

    // --- gated combine, fully coalesced emit ---
#pragma unroll
    for (int j = 0; j < 8; ++j) {
      const int fi = (j * 32 + lane) * 4;
      v4f z2v = *(const v4f*)(zT + fi);          // ds_read_b128
      const float gt = gv[j];
      v4f hn;
#pragma unroll
      for (int c = 0; c < 4; ++c) hn[c] = (1.0f - gt) * z1r[j][c] + gt * z2v[c];
      __builtin_nontemporal_store(hn, (v4f*)(op + fi));  // streamed, don't pollute L2
      *(v4bf*)(hT + fi) = cvt4_bf16(hn);         // ds_store_b64, h update
      if (t == TT - 1) *(v4f*)(hlast + rowb * 64 + fi) = hn;
    }
    asm volatile("" ::: "memory");  // keep LDS reuse ordered across iterations
  }
}

extern "C" void kernel_launch(void* const* d_in, const int* in_sizes, int n_in,
                              void* d_out, int out_size, void* d_ws, size_t ws_size,
                              hipStream_t stream) {
  (void)in_sizes; (void)n_in; (void)out_size; (void)d_ws; (void)ws_size;
  const float* x   = (const float*)d_in[0];
  const float* h   = (const float*)d_in[1];
  const float* g   = (const float*)d_in[2];
  const float* a   = (const float*)d_in[3];
  const float* Wc1 = (const float*)d_in[4];
  const float* bc1 = (const float*)d_in[5];
  const float* Wc2 = (const float*)d_in[6];
  const float* bc2 = (const float*)d_in[7];
  const float* Wp1 = (const float*)d_in[8];
  const float* bp1 = (const float*)d_in[9];
  const float* Wp2 = (const float*)d_in[10];
  const float* bp2 = (const float*)d_in[11];

  float* outs  = (float*)d_out;                    // [T,B,HID]
  float* hlast = outs + (long)TT * BB * HID;       // [B,HID]
  float* capt  = hlast + (long)BB * HID;           // [T,B,HID]

  // Parallel capture path (also produces `capt` output, consumed by the scan).
  capture_kernel<<<1024, 256, 0, stream>>>(x, Wc1, bc1, Wc2, bc2, capt);
  // Sequential scan over T, parallel over B (256 wave-tiles).
  scan_kernel<<<BB / 128, 256, 0, stream>>>(h, g, a, Wp1, bp1, Wp2, bp2,
                                            capt, outs, hlast);
}
